// ALIGNNConv_7275674599849
// MI455X (gfx1250) — matrix operand order repaired and checked by reading, not compiled, not run on hardware
//
#include <hip/hip_runtime.h>
#include <hip/hip_bf16.h>

// ---------------------------------------------------------------------------
// Edge-gated conv (ALIGNN edge update) for MI455X / gfx1250.
//   gate = sigmoid([node[row] | node[col] | ef] @ W + b);  out = ef * gate
// GEMM M=E(640k), K=384, N=128 via v_wmma_f32_16x16x32_bf16 (wave32).
// Block = 128 threads (4 waves), 128 edges/block; each wave owns 2 M-tiles
// (32 edge rows) x 8 N-tiles so every B-fragment load feeds 2 WMMAs.
// ---------------------------------------------------------------------------

typedef __attribute__((ext_vector_type(16))) __bf16 v16bf;
typedef __attribute__((ext_vector_type(8)))  float  v8f;

#define ROWSTRIDE 195   // dwords per staged LDS row: 384 bf16 = 192 dw + 3 pad
                        // odd stride -> A-frag reads are bank-conflict free

__device__ __forceinline__ unsigned int bf16_rne(float f) {
    unsigned int u = __float_as_uint(f);
    return (u + 0x7FFFu + ((u >> 16) & 1u)) >> 16;   // round-to-nearest-even
}
__device__ __forceinline__ unsigned int pack2(float a, float b) {
    return bf16_rne(a) | (bf16_rne(b) << 16);
}

// ---------------------------------------------------------------------------
// Prep: W[384][128] f32 -> bf16, pre-swizzled into WMMA B-fragment order.
// B tile (kt,nt) is 32(K) x 16(N) bf16. Layout used here (contiguous K, not the ISA table order):
//   lane = n + 16*half; dword r packs K = 2r,2r+1 (+16 for half=1).
// Stored lane-major: Wb[((kt*8+nt)*32 + lane)*8 + r]  -> 1KB contiguous/wave.
// Total 12*8*32*8 dwords = 96KB in d_ws.
// ---------------------------------------------------------------------------
__global__ void wprep_kernel(const float* __restrict__ W,
                             unsigned int* __restrict__ Wb) {
    int t = blockIdx.x * 256 + threadIdx.x;      // 0 .. 24575
    if (t >= 12 * 8 * 32 * 8) return;
    int r    =  t        & 7;
    int lane = (t >> 3)  & 31;
    int nt   = (t >> 8)  & 7;
    int kt   =  t >> 11;
    int half = lane >> 4;
    int n    = nt * 16 + (lane & 15);
    int k    = kt * 32 + 2 * r + 16 * half;
    float f0 = W[k * 128 + n];
    float f1 = W[(k + 1) * 128 + n];
    Wb[t] = pack2(f0, f1);
}

// ---------------------------------------------------------------------------
// Main kernel: 4 waves / block, 128 edges / block, N=128 full width.
// ---------------------------------------------------------------------------
__global__ __launch_bounds__(128) void edge_gate_kernel(
    const float*        __restrict__ node,   // [N_NODES][128]
    const int*          __restrict__ eidx,   // [2][E]
    const float*        __restrict__ ef,     // [E][128]
    const unsigned int* __restrict__ Wb,     // swizzled bf16 W fragments
    const float*        __restrict__ bias,   // [128]
    float*              __restrict__ out,    // [E][128]
    int nEdges)
{
    __shared__ unsigned int Xs[128 * ROWSTRIDE];  // 97.5 KB staged bf16 input

    const int lane = threadIdx.x & 31;
    const int wave = threadIdx.x >> 5;
    const long long ebase = (long long)blockIdx.x * 128;

    // ---- Stage 128 edges: gather nodes, convert f32->bf16, write LDS ----
    for (int e = wave; e < 128; e += 4) {
        long long eg = ebase + e;
        long long egc = eg < (long long)nEdges ? eg : (long long)nEdges - 1;
        int r = eidx[egc];
        int c = eidx[(long long)nEdges + egc];
        const float4 nr = ((const float4*)(node + (long long)r * 128))[lane];
        const float4 nc = ((const float4*)(node + (long long)c * 128))[lane];
        const float4 fe = ((const float4*)(ef + egc * 128))[lane];
        unsigned int base = e * ROWSTRIDE;
        Xs[base       + lane * 2    ] = pack2(nr.x, nr.y);
        Xs[base       + lane * 2 + 1] = pack2(nr.z, nr.w);
        Xs[base +  64 + lane * 2    ] = pack2(nc.x, nc.y);
        Xs[base +  64 + lane * 2 + 1] = pack2(nc.z, nc.w);
        Xs[base + 128 + lane * 2    ] = pack2(fe.x, fe.y);
        Xs[base + 128 + lane * 2 + 1] = pack2(fe.z, fe.w);
    }
    __syncthreads();

    // ---- GEMM: each wave owns 32 edge rows (2 M-tiles) x all 8 N-tiles ----
    v8f acc0[8], acc1[8];
    #pragma unroll
    for (int i = 0; i < 8; ++i) {
        acc0[i] = (v8f){0,0,0,0,0,0,0,0};
        acc1[i] = (v8f){0,0,0,0,0,0,0,0};
    }

    const int m    = lane & 15;   // A: matrix row within tile; C/D: column N
    const int half = lane >> 4;
    const unsigned rb0 = (unsigned)(wave * 32 +      m) * ROWSTRIDE;
    const unsigned rb1 = (unsigned)(wave * 32 + 16 + m) * ROWSTRIDE;

    for (int kt = 0; kt < 12; ++kt) {
        // A fragment (16x32 bf16): lane m+16*half, dword r holds
        //   K-word = r + 8*half  within this kt tile (matches Wb).
        union { v16bf v; unsigned int u[8]; } A0, A1;
        unsigned kb0 = rb0 + kt * 16 + 8 * half;
        unsigned kb1 = rb1 + kt * 16 + 8 * half;
        #pragma unroll
        for (int r = 0; r < 8; ++r) {
            A0.u[r] = Xs[kb0 + r];
            A1.u[r] = Xs[kb1 + r];
        }

        const unsigned int* wptr = Wb + (unsigned)kt * 2048 + lane * 8;
        #pragma unroll
        for (int nt = 0; nt < 8; ++nt) {
            union { v16bf v; uint4 q[2]; } B;
            const uint4* bp = (const uint4*)(wptr + nt * 256);
            B.q[0] = bp[0];
            B.q[1] = bp[1];
            acc0[nt] = __builtin_amdgcn_wmma_f32_16x16x32_bf16(
                false, A0.v, false, B.v, (short)0, acc0[nt], false, false);
            acc1[nt] = __builtin_amdgcn_wmma_f32_16x16x32_bf16(
                false, A1.v, false, B.v, (short)0, acc1[nt], false, false);
        }
    }

    // ---- Epilogue: bias + sigmoid + gate * edge_features ----
    // C/D layout: lane L -> N = L&15, VGPR v -> M = v + 8*(L>>4).
    #pragma unroll
    for (int mt = 0; mt < 2; ++mt) {
        const v8f* acc = mt ? acc1 : acc0;
        #pragma unroll
        for (int nt = 0; nt < 8; ++nt) {
            int n = nt * 16 + m;
            float bn = bias[n];
            #pragma unroll
            for (int v = 0; v < 8; ++v) {
                int eloc = wave * 32 + mt * 16 + v + 8 * half;
                long long eg = ebase + eloc;
                if (eg < (long long)nEdges) {
                    float x = acc[nt][v] + bn;
                    float g = 1.0f / (1.0f + __expf(-x));
                    out[eg * 128 + n] = ef[eg * 128 + n] * g;
                }
            }
        }
    }
}

extern "C" void kernel_launch(void* const* d_in, const int* in_sizes, int n_in,
                              void* d_out, int out_size, void* d_ws, size_t ws_size,
                              hipStream_t stream) {
    const float* node = (const float*)d_in[0];
    const int*   eidx = (const int*)  d_in[1];
    const float* ef   = (const float*)d_in[2];
    const float* W    = (const float*)d_in[3];
    const float* bias = (const float*)d_in[4];
    float* out = (float*)d_out;

    int nEdges = in_sizes[2] / 128;          // edge_features is [E][128]
    unsigned int* Wb = (unsigned int*)d_ws;  // 96 KB swizzled bf16 weights

    hipLaunchKernelGGL(wprep_kernel, dim3(96), dim3(256), 0, stream, W, Wb);

    int nBlocks = (nEdges + 127) / 128;
    hipLaunchKernelGGL(edge_gate_kernel, dim3(nBlocks), dim3(128), 0, stream,
                       node, eidx, ef, Wb, bias, out, nEdges);
}
